// AttentionCritic_64115271795028
// MI455X (gfx1250) — matrix operand order repaired and checked by reading, not compiled, not run on hardware
//
#include <hip/hip_runtime.h>

#define NA_ 20
#define NB_ 4096
#define NS_ 128
#define NAD_ 16
#define NH_ 128
#define NKH_ 4
#define ND_ 32
#define NC_ 4

typedef __attribute__((ext_vector_type(16))) _Float16 v16h;
typedef __attribute__((ext_vector_type(8)))  float    v8f;

// ---------------- BatchNorm statistics (mean, rstd) over batch axis ----------
__global__ __launch_bounds__(256)
void ac_bn_stats(const float* __restrict__ states, const float* __restrict__ actions,
                 float* __restrict__ meanA, float* __restrict__ rstdA,
                 float* __restrict__ meanC, float* __restrict__ rstdC) {
    int idx = blockIdx.x;
    int tid = threadIdx.x;
    bool isC = idx >= NA_ * 144;
    int rel  = isC ? idx - NA_ * 144 : idx;
    int grp  = rel / 144;
    int f    = rel % 144;
    float s1 = 0.f, s2 = 0.f;
    for (int b = tid; b < NB_; b += 256) {
        float v;
        if (!isC) {
            v = (f < NS_) ? states[((size_t)grp * NB_ + b) * NS_ + f]
                          : actions[((size_t)grp * NB_ + b) * NAD_ + (f - NS_)];
        } else {
            v = 0.f;
            for (int j = 0; j < 5; ++j) {
                int a = grp * 5 + j;
                v += (f < NS_) ? states[((size_t)a * NB_ + b) * NS_ + f]
                               : actions[((size_t)a * NB_ + b) * NAD_ + (f - NS_)];
            }
            v *= 0.2f;
        }
        s1 += v; s2 += v * v;
    }
    __shared__ float r1[256], r2[256];
    r1[tid] = s1; r2[tid] = s2; __syncthreads();
    for (int st = 128; st > 0; st >>= 1) {
        if (tid < st) { r1[tid] += r1[tid + st]; r2[tid] += r2[tid + st]; }
        __syncthreads();
    }
    if (tid == 0) {
        float m   = r1[0] * (1.0f / NB_);
        float var = r2[0] * (1.0f / NB_) - m * m;
        float r   = rsqrtf(var + 1e-5f);
        if (!isC) { meanA[rel] = m; rstdA[rel] = r; }
        else      { meanC[rel] = m; rstdC[rel] = r; }
    }
}

// ---------------- normalized f16 inputs (padded K: 144 -> 160) ---------------
__global__ __launch_bounds__(256)
void ac_prep_agent(const float* __restrict__ states, const float* __restrict__ actions,
                   const float* __restrict__ meanA, const float* __restrict__ rstdA,
                   _Float16* __restrict__ Xsa, _Float16* __restrict__ Xse) {
    long i = (long)blockIdx.x * 256 + threadIdx.x;
    if (i >= (long)NA_ * NB_ * 160) return;
    int  f   = (int)(i % 160);
    long rem = i / 160;
    int  b   = (int)(rem % NB_);
    int  a   = (int)(rem / NB_);
    float v = 0.f;
    if (f < 144) {
        float raw = (f < NS_) ? states[((size_t)a * NB_ + b) * NS_ + f]
                              : actions[((size_t)a * NB_ + b) * NAD_ + (f - NS_)];
        v = (raw - meanA[a * 144 + f]) * rstdA[a * 144 + f];
    }
    Xsa[i] = (_Float16)v;
    if (f < NS_) Xse[((size_t)a * NB_ + b) * NS_ + f] = (_Float16)v;
}

__global__ __launch_bounds__(256)
void ac_prep_cluster(const float* __restrict__ states, const float* __restrict__ actions,
                     const float* __restrict__ meanC, const float* __restrict__ rstdC,
                     _Float16* __restrict__ Xcsa, _Float16* __restrict__ Xcse) {
    long i = (long)blockIdx.x * 256 + threadIdx.x;
    if (i >= (long)NC_ * NB_ * 160) return;
    int  f   = (int)(i % 160);
    long rem = i / 160;
    int  b   = (int)(rem % NB_);
    int  c   = (int)(rem / NB_);
    float v = 0.f;
    if (f < 144) {
        float raw = 0.f;
        for (int j = 0; j < 5; ++j) {
            int a = c * 5 + j;
            raw += (f < NS_) ? states[((size_t)a * NB_ + b) * NS_ + f]
                             : actions[((size_t)a * NB_ + b) * NAD_ + (f - NS_)];
        }
        raw *= 0.2f;
        v = (raw - meanC[c * 144 + f]) * rstdC[c * 144 + f];
    }
    Xcsa[i] = (_Float16)v;
    if (f < NS_) Xcse[((size_t)c * NB_ + b) * NS_ + f] = (_Float16)v;
}

// -- weight convert: src [G][Ks][N] f32 -> dst [G][Npad][Kpad] f16 (zero pad) --
__global__ __launch_bounds__(256)
void ac_conv_wt(const float* __restrict__ src, _Float16* __restrict__ dst,
                int G, int Ks, int Kpad, int N, int Npad) {
    long i = (long)blockIdx.x * 256 + threadIdx.x;
    if (i >= (long)G * Npad * Kpad) return;
    int  k   = (int)(i % Kpad);
    long rem = i / Kpad;
    int  n   = (int)(rem % Npad);
    int  g   = (int)(rem / Npad);
    float v = (k < Ks && n < N) ? src[((size_t)g * Ks + k) * N + n] : 0.f;
    dst[i] = (_Float16)v;
}

// ---------------- WMMA fragment loads (ISA 7.12.2 layouts) -------------------
__device__ __forceinline__ v16h ac_ldfragA(const _Float16* __restrict__ row, int k0, int half) {
    union { v16h v; unsigned u[8]; } f;
#pragma unroll
    for (int t = 0; t < 8; ++t) {
        int ka = k0 + ((t >> 2) << 4) + (half << 3) + ((t & 3) << 1);
        f.u[t] = *(const unsigned*)(row + ka);
    }
    return f.v;
}
__device__ __forceinline__ v16h ac_ldfragB(const _Float16* __restrict__ row, int k0, int half) {
    union { v16h v; unsigned u[8]; } f;
#pragma unroll
    for (int t = 0; t < 8; ++t)
        f.u[t] = *(const unsigned*)(row + k0 + ((t >> 2) << 4) + (half << 3) + ((t & 3) << 1));
    return f.v;
}

// ---------------- WMMA GEMM: Y[g] = act(X[xg] @ W[wg]^T + bias[wg]) ----------
// Register-blocked: each wave owns a 32x32 macro-tile = 2 A-frags x 2 B-frags
// -> 4 v_wmma per 4 fragment loads (2x VMEM traffic reduction vs 16x16/wave).
// cw2 = column-waves per workgroup (2 when N>=64 else 1); rowWaves = 8/cw2.
__global__ __launch_bounds__(256)
void ac_gemm_wmma(const _Float16* __restrict__ X, int xstride, long xgstride,
                  const _Float16* __restrict__ W, const float* __restrict__ bias,
                  float* __restrict__ Y32, _Float16* __restrict__ Y16,
                  int ystride16, int yoff16,
                  int Kd, int N, int Npad, int xmod, int wdiv, int act, int cw2) {
    int g    = blockIdx.z;
    int xg   = g % xmod;
    int wg   = g / wdiv;
    int wave = threadIdx.x >> 5;
    int lane = threadIdx.x & 31;
    int rowWaves = 8 / cw2;
    int rw   = wave % rowWaves;
    int cw   = wave / rowWaves;
    int row0 = (blockIdx.x * rowWaves + rw) * 32;
    int col0 = (blockIdx.y * cw2 + cw) * 32;
    int mr   = lane & 15;
    int half = lane >> 4;
    const _Float16* xr0 = X + (long)xg * xgstride + (long)(row0 + mr) * xstride;
    const _Float16* xr1 = xr0 + (long)16 * xstride;
    const _Float16* wr0 = W + (long)wg * Npad * Kd + (long)(col0 + mr) * Kd;
    const _Float16* wr1 = wr0 + (long)16 * Kd;
    v8f a00 = {}, a01 = {}, a10 = {}, a11 = {};
    for (int k0 = 0; k0 < Kd; k0 += 32) {
        v16h fa0 = ac_ldfragA(xr0, k0, half);
        v16h fa1 = ac_ldfragA(xr1, k0, half);
        v16h fb0 = ac_ldfragB(wr0, k0, half);
        v16h fb1 = ac_ldfragB(wr1, k0, half);
        a00 = __builtin_amdgcn_wmma_f32_16x16x32_f16(false, fa0, false, fb0, (short)0, a00, false, false);
        a01 = __builtin_amdgcn_wmma_f32_16x16x32_f16(false, fa0, false, fb1, (short)0, a01, false, false);
        a10 = __builtin_amdgcn_wmma_f32_16x16x32_f16(false, fa1, false, fb0, (short)0, a10, false, false);
        a11 = __builtin_amdgcn_wmma_f32_16x16x32_f16(false, fa1, false, fb1, (short)0, a11, false, false);
    }
    int n0 = col0 + mr;
    int n1 = col0 + 16 + mr;
    float bv0 = (bias && n0 < N) ? bias[wg * N + n0] : 0.f;
    float bv1 = (bias && n1 < N) ? bias[wg * N + n1] : 0.f;
    long yb = (long)g * NB_;
#pragma unroll
    for (int t = 0; t < 8; ++t) {
        int r0 = row0 + half * 8 + t;
        int r1 = r0 + 16;
        float v00 = a00[t] + bv0, v01 = a01[t] + bv1;
        float v10 = a10[t] + bv0, v11 = a11[t] + bv1;
        if (act) {
            v00 = v00 > 0.f ? v00 : 0.01f * v00;
            v01 = v01 > 0.f ? v01 : 0.01f * v01;
            v10 = v10 > 0.f ? v10 : 0.01f * v10;
            v11 = v11 > 0.f ? v11 : 0.01f * v11;
        }
        if (n0 < N) {
            if (Y32) { Y32[(yb + r0) * (long)N + n0] = v00; Y32[(yb + r1) * (long)N + n0] = v10; }
            if (Y16) {
                Y16[(yb + r0) * (long)ystride16 + yoff16 + n0] = (_Float16)v00;
                Y16[(yb + r1) * (long)ystride16 + yoff16 + n0] = (_Float16)v10;
            }
        }
        if (n1 < N) {
            if (Y32) { Y32[(yb + r0) * (long)N + n1] = v01; Y32[(yb + r1) * (long)N + n1] = v11; }
            if (Y16) {
                Y16[(yb + r0) * (long)ystride16 + yoff16 + n1] = (_Float16)v01;
                Y16[(yb + r1) * (long)ystride16 + yoff16 + n1] = (_Float16)v11;
            }
        }
    }
}

// ---------------- attention: softmax((sel . key)/sqrt(D)) @ vals -------------
// keys/sels/vals: [K][NAg][B][D] f16. One workgroup per batch element.
__global__ __launch_bounds__(256)
void ac_attention(const _Float16* __restrict__ keys, const _Float16* __restrict__ sels,
                  const _Float16* __restrict__ vals, int NAg,
                  float* __restrict__ out32, _Float16* __restrict__ out16,
                  int o16_stride, int o16_off) {
    int b = blockIdx.x;
    __shared__ float sk[NKH_ * NA_ * ND_];
    __shared__ float ss[NKH_ * NA_ * ND_];
    __shared__ float sv[NKH_ * NA_ * ND_];
    __shared__ float sp[NKH_ * NA_ * NA_];
    int tot = NKH_ * NAg * ND_;
    for (int i = threadIdx.x; i < tot; i += 256) {
        int k  = i / (NAg * ND_);
        int rm = i % (NAg * ND_);
        int a  = rm / ND_;
        int d  = rm % ND_;
        size_t gi = ((size_t)(k * NAg + a) * NB_ + b) * ND_ + d;
        sk[i] = (float)keys[gi];
        ss[i] = (float)sels[gi];
        sv[i] = (float)vals[gi];
    }
    __syncthreads();
    for (int p = threadIdx.x; p < NKH_ * NAg; p += 256) {
        int k = p / NAg, ii = p % NAg;
        float logit[NA_];
        float mx = -1e30f;
        for (int j = 0; j < NAg; ++j) {
            float acc = 0.f;
            for (int d = 0; d < ND_; ++d)
                acc += ss[(k * NAg + ii) * ND_ + d] * sk[(k * NAg + j) * ND_ + d];
            acc *= 0.17677669529663688f;          // 1/sqrt(32)
            if (j == ii) acc = -1e9f;
            logit[j] = acc;
            mx = fmaxf(mx, acc);
        }
        float sum = 0.f;
        for (int j = 0; j < NAg; ++j) { float e = expf(logit[j] - mx); logit[j] = e; sum += e; }
        float inv = 1.f / sum;
        for (int j = 0; j < NAg; ++j) sp[(k * NAg + ii) * NAg + j] = logit[j] * inv;
    }
    __syncthreads();
    for (int i = threadIdx.x; i < tot; i += 256) {
        int k  = i / (NAg * ND_);
        int rm = i % (NAg * ND_);
        int a  = rm / ND_;
        int d  = rm % ND_;
        float acc = 0.f;
        for (int j = 0; j < NAg; ++j)
            acc += sp[(k * NAg + a) * NAg + j] * sv[(k * NAg + j) * ND_ + d];
        int col = k * ND_ + d;
        if (out32) out32[((size_t)a * NB_ + b) * NH_ + col] = acc;
        if (out16) out16[((size_t)a * NB_ + b) * o16_stride + o16_off + col] = (_Float16)acc;
    }
}

extern "C" void kernel_launch(void* const* d_in, const int* in_sizes, int n_in,
                              void* d_out, int out_size, void* d_ws, size_t ws_size,
                              hipStream_t stream) {
    (void)in_sizes; (void)n_in; (void)out_size; (void)ws_size;
    const float* states    = (const float*)d_in[0];
    const float* actions   = (const float*)d_in[1];
    const float* enc_W     = (const float*)d_in[2];
    const float* enc_b     = (const float*)d_in[3];
    const float* senc_W    = (const float*)d_in[4];
    const float* senc_b    = (const float*)d_in[5];
    const float* key_W     = (const float*)d_in[6];
    const float* sel_W     = (const float*)d_in[7];
    const float* val_W     = (const float*)d_in[8];
    const float* val_b     = (const float*)d_in[9];
    const float* cenc_W    = (const float*)d_in[10];
    const float* cenc_b    = (const float*)d_in[11];
    const float* csenc_W   = (const float*)d_in[12];
    const float* csenc_b   = (const float*)d_in[13];
    const float* ckey_W    = (const float*)d_in[14];
    const float* csel_W    = (const float*)d_in[15];
    const float* cval_W    = (const float*)d_in[16];
    const float* cval_b    = (const float*)d_in[17];
    const float* critic_W1 = (const float*)d_in[18];
    const float* critic_b1 = (const float*)d_in[19];
    const float* critic_W2 = (const float*)d_in[20];
    const float* critic_b2 = (const float*)d_in[21];

    float* out_q     = (float*)d_out;                         // [A,B,AD]
    float* out_cattn = out_q + (size_t)NA_ * NB_ * NAD_;      // [C,B,H]

    char* p = (char*)d_ws;
    auto allocB = [&](size_t bytes) { char* r = p; p += (bytes + 255) & ~(size_t)255; return r; };
    auto allocF = [&](size_t n) { return (float*)allocB(n * sizeof(float)); };
    auto allocH = [&](size_t n) { return (_Float16*)allocB(n * sizeof(_Float16)); };

    float* meanA = allocF(NA_ * 144); float* rstdA = allocF(NA_ * 144);
    float* meanC = allocF(NC_ * 144); float* rstdC = allocF(NC_ * 144);

    _Float16* Xsa   = allocH((size_t)NA_ * NB_ * 160);
    _Float16* Xse   = allocH((size_t)NA_ * NB_ * 128);
    _Float16* Xcsa  = allocH((size_t)NC_ * NB_ * 160);
    _Float16* Xcse  = allocH((size_t)NC_ * NB_ * 128);
    _Float16* sa16  = allocH((size_t)NA_ * NB_ * 128);
    _Float16* csa16 = allocH((size_t)NC_ * NB_ * 128);
    _Float16* cse16 = allocH((size_t)NC_ * NB_ * 128);
    _Float16* Xcrit = allocH((size_t)NA_ * NB_ * 256);   // [se | attn] f16
    _Float16* h16   = allocH((size_t)NA_ * NB_ * 128);
    _Float16* keys16 = allocH((size_t)NKH_ * NA_ * NB_ * 32);
    _Float16* sels16 = allocH((size_t)NKH_ * NA_ * NB_ * 32);
    _Float16* vals16 = allocH((size_t)NKH_ * NA_ * NB_ * 32);
    _Float16* ck16   = allocH((size_t)NKH_ * NC_ * NB_ * 32);
    _Float16* cs16   = allocH((size_t)NKH_ * NC_ * NB_ * 32);
    _Float16* cv16   = allocH((size_t)NKH_ * NC_ * NB_ * 32);
    _Float16* encWt   = allocH((size_t)NA_ * 128 * 160);
    _Float16* sencWt  = allocH((size_t)NA_ * 128 * 128);
    _Float16* keyWt   = allocH((size_t)NKH_ * 32 * 128);
    _Float16* selWt   = allocH((size_t)NKH_ * 32 * 128);
    _Float16* valWt   = allocH((size_t)NKH_ * 32 * 128);
    _Float16* cencWt  = allocH((size_t)NC_ * 128 * 160);
    _Float16* csencWt = allocH((size_t)NC_ * 128 * 128);
    _Float16* ckeyWt  = allocH((size_t)NKH_ * 32 * 128);
    _Float16* cselWt  = allocH((size_t)NKH_ * 32 * 128);
    _Float16* cvalWt  = allocH((size_t)NKH_ * 32 * 128);
    _Float16* critW1t = allocH((size_t)NA_ * 128 * 256);
    _Float16* critW2t = allocH((size_t)NA_ * 32 * 128);  // N=16 padded to 32

    // 1) BN statistics
    ac_bn_stats<<<(NA_ + NC_) * 144, 256, 0, stream>>>(states, actions, meanA, rstdA, meanC, rstdC);

    // 2) normalized f16 inputs
    {
        long tot = (long)NA_ * NB_ * 160;
        ac_prep_agent<<<(unsigned)((tot + 255) / 256), 256, 0, stream>>>(states, actions, meanA, rstdA, Xsa, Xse);
        tot = (long)NC_ * NB_ * 160;
        ac_prep_cluster<<<(unsigned)((tot + 255) / 256), 256, 0, stream>>>(states, actions, meanC, rstdC, Xcsa, Xcse);
    }

    // 3) weights -> transposed, padded f16
    auto conv = [&](const float* src, _Float16* dst, int G, int Ks, int Kpad, int N, int Npad) {
        long tot = (long)G * Npad * Kpad;
        ac_conv_wt<<<(unsigned)((tot + 255) / 256), 256, 0, stream>>>(src, dst, G, Ks, Kpad, N, Npad);
    };
    conv(enc_W,     encWt,   NA_, 144, 160, 128, 128);
    conv(senc_W,    sencWt,  NA_, 128, 128, 128, 128);
    conv(key_W,     keyWt,   NKH_, 128, 128, 32, 32);
    conv(sel_W,     selWt,   NKH_, 128, 128, 32, 32);
    conv(val_W,     valWt,   NKH_, 128, 128, 32, 32);
    conv(cenc_W,    cencWt,  NC_, 144, 160, 128, 128);
    conv(csenc_W,   csencWt, NC_, 128, 128, 128, 128);
    conv(ckey_W,    ckeyWt,  NKH_, 128, 128, 32, 32);
    conv(csel_W,    cselWt,  NKH_, 128, 128, 32, 32);
    conv(cval_W,    cvalWt,  NKH_, 128, 128, 32, 32);
    conv(critic_W1, critW1t, NA_, 256, 256, 128, 128);
    conv(critic_W2, critW2t, NA_, 128, 128, 16, 32);

    // 4) WMMA GEMMs (register-blocked 32x32 per wave)
    auto gemm = [&](const _Float16* X, int xstride, const _Float16* W, const float* bias,
                    float* Y32, _Float16* Y16, int ys, int yo,
                    int Kd, int N, int Npad, int G, int xmod, int wdiv, int act) {
        int cw2 = (N >= 64) ? 2 : 1;
        int rowWaves = 8 / cw2;
        dim3 grid((unsigned)(NB_ / (32 * rowWaves)), (unsigned)((N + 32 * cw2 - 1) / (32 * cw2)), (unsigned)G);
        ac_gemm_wmma<<<grid, 256, 0, stream>>>(X, xstride, (long)NB_ * xstride, W, bias,
                                               Y32, Y16, ys, yo, Kd, N, Npad, xmod, wdiv, act, cw2);
    };
    // encoders (LeakyReLU); se goes f16 straight into Xcrit[:,:, 0:128]
    gemm(Xsa,  160, encWt,   enc_b,   nullptr, sa16,  128, 0, 160, 128, 128, NA_, NA_, 1, 1);
    gemm(Xse,  128, sencWt,  senc_b,  nullptr, Xcrit, 256, 0, 128, 128, 128, NA_, NA_, 1, 1);
    gemm(Xcsa, 160, cencWt,  cenc_b,  nullptr, csa16, 128, 0, 160, 128, 128, NC_, NC_, 1, 1);
    gemm(Xcse, 128, csencWt, csenc_b, nullptr, cse16, 128, 0, 128, 128, 128, NC_, NC_, 1, 1);
    // per-head projections: g = k*NAg + a
    gemm(sa16,  128, keyWt,  nullptr, nullptr, keys16, 32, 0, 128, 32, 32, NKH_ * NA_, NA_, NA_, 0);
    gemm(Xcrit, 256, selWt,  nullptr, nullptr, sels16, 32, 0, 128, 32, 32, NKH_ * NA_, NA_, NA_, 0);
    gemm(sa16,  128, valWt,  val_b,   nullptr, vals16, 32, 0, 128, 32, 32, NKH_ * NA_, NA_, NA_, 1);
    gemm(csa16, 128, ckeyWt, nullptr, nullptr, ck16,   32, 0, 128, 32, 32, NKH_ * NC_, NC_, NC_, 0);
    gemm(cse16, 128, cselWt, nullptr, nullptr, cs16,   32, 0, 128, 32, 32, NKH_ * NC_, NC_, NC_, 0);
    gemm(csa16, 128, cvalWt, cval_b,  nullptr, cv16,   32, 0, 128, 32, 32, NKH_ * NC_, NC_, NC_, 1);

    // 5) attention: agents -> Xcrit[:,:,128:256] (f16); clusters -> d_out (f32)
    ac_attention<<<NB_, 256, 0, stream>>>(keys16, sels16, vals16, NA_, nullptr, Xcrit, 256, 128);
    ac_attention<<<NB_, 256, 0, stream>>>(ck16, cs16, cv16, NC_, out_cattn, nullptr, 0, 0);

    // 6) critic
    gemm(Xcrit, 256, critW1t, critic_b1, nullptr, h16, 128, 0, 256, 128, 128, NA_, NA_, 1, 1);
    gemm(h16,   128, critW2t, critic_b2, out_q, nullptr, 0, 0, 128, 16, 32, NA_, NA_, 1, 0);
}
